// Critic_35596688949271
// MI455X (gfx1250) — matrix operand order repaired and checked by reading, not compiled, not run on hardware
//
#include <hip/hip_runtime.h>
#include <hip/hip_bf16.h>
#include <math.h>

// ---------------- problem constants ----------------
#define NB    512
#define LQ    640
#define LEADS 3
#define T1    322
#define C1    64
#define T2    161
#define C2    128
#define T3    79
#define C3    256
#define K2    (C1*7)      // 448
#define K3    (C2*9)      // 1152
#define M2    (NB*T2)     // 82432
#define M3    (NB*T3)     // 40448
#define FLATD 20224       // 256*79
#define NMB_R 500
#define NMB_P 512

typedef __attribute__((ext_vector_type(16))) __bf16 v16bf;
typedef __attribute__((ext_vector_type(8)))  float  v8f;
typedef __attribute__((ext_vector_type(8)))  unsigned int v8u;

union FragU {
    v8u   u;
    v16bf b;
    uint4 q[2];
};

static __device__ __forceinline__ unsigned short f2bf(float f) {
    union { float f; unsigned u; } v; v.f = f;
    unsigned r = v.u + 0x7FFFu + ((v.u >> 16) & 1u);   // round-to-nearest-even
    return (unsigned short)(r >> 16);
}

// ---------------- spectral norm sigma: scale = 1/||W normalize(W^T u)|| ----------------
__global__ void sn_sigma_kernel(const float* __restrict__ w, const float* __restrict__ u,
                                int O, int F, float* __restrict__ scaleOut) {
    __shared__ float sv[1200];
    __shared__ float red[256];
    int tid = threadIdx.x;
    for (int j = tid; j < F; j += 256) {
        float acc = 0.f;
        for (int o = 0; o < O; ++o) acc += w[o*F + j] * u[o];
        sv[j] = acc;
    }
    __syncthreads();
    float loc = 0.f;
    for (int j = tid; j < F; j += 256) loc += sv[j] * sv[j];
    red[tid] = loc; __syncthreads();
    for (int s = 128; s > 0; s >>= 1) { if (tid < s) red[tid] += red[tid + s]; __syncthreads(); }
    float vnorm = sqrtf(red[0]) + 1e-12f;
    __syncthreads();
    loc = 0.f;
    for (int o = tid; o < O; o += 256) {
        float acc = 0.f;
        for (int j = 0; j < F; ++j) acc += w[o*F + j] * sv[j];
        acc /= vnorm;
        loc += acc * acc;
    }
    red[tid] = loc; __syncthreads();
    for (int s = 128; s > 0; s >>= 1) { if (tid < s) red[tid] += red[tid + s]; __syncthreads(); }
    if (tid == 0) {
        float tt = red[0];
        float sigma = tt / (sqrtf(tt) + 1e-12f);
        scaleOut[0] = 1.f / sigma;
    }
}

// ---------------- conv1 (3->64, k=5, s=2, p=4), fp32 direct + leaky ----------------
__global__ void conv1_kernel(const float* __restrict__ ecg, const float* __restrict__ w1,
                             const float* __restrict__ b1, const float* __restrict__ scale,
                             float* __restrict__ x1) {
    long long idx = (long long)blockIdx.x * blockDim.x + threadIdx.x;
    const long long total = (long long)NB * C1 * T1;
    if (idx >= total) return;
    int t = (int)(idx % T1);
    int o = (int)((idx / T1) % C1);
    int n = (int)(idx / ((long long)T1 * C1));
    float s = scale[0];
    float acc = 0.f;
    for (int c = 0; c < LEADS; ++c)
        for (int k = 0; k < 5; ++k) {
            int pos = 2*t + k - 4;
            if (pos >= 0 && pos < LQ)
                acc += w1[o*15 + c*5 + k] * ecg[((long long)n*LQ + pos)*LEADS + c];
        }
    float y = acc * s + b1[o];
    x1[idx] = (y >= 0.f) ? y : 0.2f * y;
}

// ---------------- scaled fp32 -> bf16 weight convert ----------------
__global__ void wcvt_kernel(const float* __restrict__ w, unsigned short* __restrict__ out,
                            const float* __restrict__ scale, int total) {
    int i = blockIdx.x * blockDim.x + threadIdx.x;
    if (i < total) out[i] = f2bf(w[i] * scale[0]);
}

__global__ void cvt_bf16_kernel(const float* __restrict__ x, unsigned short* __restrict__ out,
                                long long total) {
    long long i = (long long)blockIdx.x * blockDim.x + threadIdx.x;
    if (i < total) out[i] = f2bf(x[i]);
}

// ---------------- im2col (NCH, stride 2) fp32 -> bf16 ----------------
__global__ void im2col_kernel(const float* __restrict__ x, unsigned short* __restrict__ out,
                              int Cin, int Tin, int Tout, int ksz, int pad) {
    long long idx = (long long)blockIdx.x * blockDim.x + threadIdx.x;
    long long total = (long long)NB * Tout * Cin * ksz;
    if (idx >= total) return;
    int k = (int)(idx % ksz);
    int c = (int)((idx / ksz) % Cin);
    int t = (int)((idx / ((long long)ksz * Cin)) % Tout);
    int n = (int)(idx / ((long long)ksz * Cin * Tout));
    int pos = 2*t + k - pad;
    float v = 0.f;
    if (pos >= 0 && pos < Tin) v = x[((long long)n*Cin + c)*Tin + pos];
    out[idx] = f2bf(v);
}

// ---------------- mb_w (K x 500) -> Bt bf16 (512 x K), zero-padded cols ----------------
__global__ void mbwt_kernel(const float* __restrict__ mbw, unsigned short* __restrict__ Bt) {
    long long idx = (long long)blockIdx.x * blockDim.x + threadIdx.x;
    long long total = (long long)NMB_P * FLATD;
    if (idx >= total) return;
    int kk  = (int)(idx % FLATD);
    int col = (int)(idx / FLATD);
    Bt[idx] = (col < NMB_R) ? f2bf(mbw[(long long)kk*NMB_R + col]) : (unsigned short)0;
}

// ---------------- epilogue store for one 16x16 C tile ----------------
static __device__ __forceinline__ void epilogue_store(
        v8f acc, int rowBase, int col, int half, int M, int Nreal, int mode,
        int T, int ldc, int coutStride, const float* __restrict__ bias,
        float* __restrict__ Cout) {
    if (col >= Nreal) return;
    float bv = (mode == 1) ? bias[col] : 0.f;
    #pragma unroll
    for (int v = 0; v < 8; ++v) {
        int row = rowBase + v + half * 8;
        if (row >= M) continue;
        float val = acc[v];
        if (mode == 1) {
            val += bv;
            val = (val >= 0.f) ? val : 0.2f * val;
            int n = row / T, t = row % T;
            Cout[(size_t)n * coutStride + (size_t)col * T + t] = val;
        } else {
            Cout[(size_t)row * ldc + col] = val;
        }
    }
}

// ---------------- bf16 WMMA GEMM with async-LDS B staging -------------------------------
// C = A(MxK,row) * B(KxN), B given as Bt: N x K (column-major B).
// Block = 4 waves stacked in M -> block tile 128(M) x 64(N); wave tile 32x64
// (2 A-frags x 4 B-frags, 8 accumulators).
// B k-chunks (64 cols x 32 k, 4KB) are staged in LDS once per block via
// global_load_async_to_lds_b128 (ASYNCcnt), double-buffered, so the copy of
// chunk kc+1 overlaps the WMMAs of chunk kc. A is loaded register-direct.
// REQUIRES: M % 128 == 0, N % 64 == 0, K % 32 == 0 (holds for all call sites;
// no early-outs allowed because of the barriers).
// mode 0: C[row*ldc+col] = v               (col < Nreal guard)
// mode 1: row=(n*T+t): out[n*coutStride + col*T + t] = leaky(v + bias[col])
#define BCOLS 64
#define BKPAD 40   // 32 k-elements padded to 40 (80B col stride, 16B-aligned)
__global__ void __launch_bounds__(128)
gemm_bf16_wmma_kernel(const unsigned short* __restrict__ A,
                      const unsigned short* __restrict__ Bt,
                      const float* __restrict__ bias,
                      float* __restrict__ Cout,
                      int M, int K, int Nreal, int mode,
                      int T, int ldc, int coutStride) {
    __shared__ unsigned short bsh[2][BCOLS][BKPAD];

    const int tid  = threadIdx.x;
    const int wave = tid >> 5;                         // 4 waves / block
    const int lane = tid & 31;
    const int m0 = (blockIdx.y * 4 + wave) * 32;       // 32 rows per wave
    const int n0 = blockIdx.x * 64;                    // 64 cols per block/wave
    const int half = lane >> 4;
    const int l16  = lane & 15;

    // --- async B-stage assignment: thread -> (col, 32B segment) ---
    const int ccol = tid >> 1;                         // 0..63
    const int cseg = tid & 1;                          // 0..1 (two 32B halves of a 64B col-chunk)
    const unsigned short* gB = Bt + (size_t)(n0 + ccol) * K + cseg * 16;

    const unsigned short* Ab0 = A  + (size_t)(m0 + l16) * K + half * 16;
    const unsigned short* Ab1 = Ab0 + (size_t)16 * K;

    v8f c00 = {}, c01 = {}, c02 = {}, c03 = {};
    v8f c10 = {}, c11 = {}, c12 = {}, c13 = {};

    const int nk = K >> 5;

    // prologue: fill buffer 0 with k-chunk 0
    {
        unsigned lo = (unsigned)(unsigned long long)&bsh[0][ccol][cseg * 16];
        const unsigned short* g = gB;
        asm volatile("global_load_async_to_lds_b128 %0, %1, off\n\t"
                     "global_load_async_to_lds_b128 %0, %1, off offset:16"
                     :: "v"(lo), "v"(g) : "memory");
    }

    for (int kc = 0; kc < nk; ++kc) {
        const int cur = kc & 1;
        // fill of buffer `cur` (issued last iteration / prologue) is complete after this:
        asm volatile("s_wait_asynccnt 0x0" ::: "memory");
        __syncthreads();
        // prefetch next chunk into the other buffer (its prior readers are past the barrier)
        if (kc + 1 < nk) {
            unsigned lo = (unsigned)(unsigned long long)&bsh[cur ^ 1][ccol][cseg * 16];
            const unsigned short* g = gB + (size_t)(kc + 1) * 32;
            asm volatile("global_load_async_to_lds_b128 %0, %1, off\n\t"
                         "global_load_async_to_lds_b128 %0, %1, off offset:16"
                         :: "v"(lo), "v"(g) : "memory");
        }

        const int k0 = kc * 32;
        FragU a0, a1, b0, b1, b2, b3;
        a0.q[0] = *(const uint4*)(Ab0 + k0); a0.q[1] = *(const uint4*)(Ab0 + k0 + 8);
        a1.q[0] = *(const uint4*)(Ab1 + k0); a1.q[1] = *(const uint4*)(Ab1 + k0 + 8);
        {
            const uint4* l0 = (const uint4*)&bsh[cur][ 0 + l16][half * 16];
            const uint4* l1 = (const uint4*)&bsh[cur][16 + l16][half * 16];
            const uint4* l2 = (const uint4*)&bsh[cur][32 + l16][half * 16];
            const uint4* l3 = (const uint4*)&bsh[cur][48 + l16][half * 16];
            b0.q[0] = l0[0]; b0.q[1] = l0[1];
            b1.q[0] = l1[0]; b1.q[1] = l1[1];
            b2.q[0] = l2[0]; b2.q[1] = l2[1];
            b3.q[0] = l3[0]; b3.q[1] = l3[1];
        }

        c00 = __builtin_amdgcn_wmma_f32_16x16x32_bf16(false, a0.b, false, b0.b, (short)0, c00, false, false);
        c01 = __builtin_amdgcn_wmma_f32_16x16x32_bf16(false, a0.b, false, b1.b, (short)0, c01, false, false);
        c02 = __builtin_amdgcn_wmma_f32_16x16x32_bf16(false, a0.b, false, b2.b, (short)0, c02, false, false);
        c03 = __builtin_amdgcn_wmma_f32_16x16x32_bf16(false, a0.b, false, b3.b, (short)0, c03, false, false);
        c10 = __builtin_amdgcn_wmma_f32_16x16x32_bf16(false, a1.b, false, b0.b, (short)0, c10, false, false);
        c11 = __builtin_amdgcn_wmma_f32_16x16x32_bf16(false, a1.b, false, b1.b, (short)0, c11, false, false);
        c12 = __builtin_amdgcn_wmma_f32_16x16x32_bf16(false, a1.b, false, b2.b, (short)0, c12, false, false);
        c13 = __builtin_amdgcn_wmma_f32_16x16x32_bf16(false, a1.b, false, b3.b, (short)0, c13, false, false);
    }

    epilogue_store(c00, m0 +  0, n0 +  0 + l16, half, M, Nreal, mode, T, ldc, coutStride, bias, Cout);
    epilogue_store(c01, m0 +  0, n0 + 16 + l16, half, M, Nreal, mode, T, ldc, coutStride, bias, Cout);
    epilogue_store(c02, m0 +  0, n0 + 32 + l16, half, M, Nreal, mode, T, ldc, coutStride, bias, Cout);
    epilogue_store(c03, m0 +  0, n0 + 48 + l16, half, M, Nreal, mode, T, ldc, coutStride, bias, Cout);
    epilogue_store(c10, m0 + 16, n0 +  0 + l16, half, M, Nreal, mode, T, ldc, coutStride, bias, Cout);
    epilogue_store(c11, m0 + 16, n0 + 16 + l16, half, M, Nreal, mode, T, ldc, coutStride, bias, Cout);
    epilogue_store(c12, m0 + 16, n0 + 32 + l16, half, M, Nreal, mode, T, ldc, coutStride, bias, Cout);
    epilogue_store(c13, m0 + 16, n0 + 48 + l16, half, M, Nreal, mode, T, ldc, coutStride, bias, Cout);
}

// ---------------- minibatch-discrimination features ----------------
__global__ void feats_kernel(const float* __restrict__ act, float* __restrict__ feats) {
    int idx = blockIdx.x * blockDim.x + threadIdx.x;
    if (idx >= NB * 100) return;
    int kc = idx % 100;
    int i  = idx / 100;
    const float* ai = act + (size_t)i * NMB_R + kc * 5;
    float a0 = ai[0], a1 = ai[1], a2 = ai[2], a3 = ai[3], a4 = ai[4];
    float s = 0.f;
    for (int j = 0; j < NB; ++j) {
        const float* aj = act + (size_t)j * NMB_R + kc * 5;
        float l1 = fabsf(a0 - aj[0]) + fabsf(a1 - aj[1]) + fabsf(a2 - aj[2])
                 + fabsf(a3 - aj[3]) + fabsf(a4 - aj[4]);
        s += expf(-l1);
    }
    feats[idx] = s;
}

// ---------------- condition embedding branch (COND_DIM = 1) ----------------
__global__ void cond_kernel(const int* __restrict__ condition, const float* __restrict__ emb,
                            const float* __restrict__ cond_w, const float* __restrict__ cond_b,
                            float* __restrict__ condf) {
    int idx = blockIdx.x * blockDim.x + threadIdx.x;
    if (idx >= NB * 50) return;
    int c = idx % 50;
    int n = idx / 50;
    float e = emb[condition[n]];
    condf[idx] = e * cond_w[c] + cond_b[c];
}

// ---------------- final fc over concat([flat, feats, cond]) ----------------
__global__ void fc_kernel(const float* __restrict__ flat, const float* __restrict__ feats,
                          const float* __restrict__ condf, const float* __restrict__ fcw,
                          const float* __restrict__ fcb, float* __restrict__ out) {
    __shared__ float red[256];
    int n = blockIdx.x, tid = threadIdx.x;
    float acc = 0.f;
    const float* fr = flat + (size_t)n * FLATD;
    for (int f = tid; f < FLATD; f += 256) acc += fr[f] * fcw[f];
    for (int k = tid; k < 100; k += 256) acc += feats[n*100 + k] * fcw[FLATD + k];
    for (int c = tid; c < 50;  c += 256) acc += condf[n*50 + c]  * fcw[FLATD + 100 + c];
    red[tid] = acc; __syncthreads();
    for (int s = 128; s > 0; s >>= 1) { if (tid < s) red[tid] += red[tid + s]; __syncthreads(); }
    if (tid == 0) out[n] = red[0] + fcb[0];
}

// ---------------- launcher ----------------
extern "C" void kernel_launch(void* const* d_in, const int* in_sizes, int n_in,
                              void* d_out, int out_size, void* d_ws, size_t ws_size,
                              hipStream_t stream) {
    (void)in_sizes; (void)n_in; (void)out_size; (void)ws_size;
    const float* ecg    = (const float*)d_in[0];
    const int*   condin = (const int*)  d_in[1];
    const float* w1     = (const float*)d_in[2];
    const float* b1     = (const float*)d_in[3];
    const float* u1     = (const float*)d_in[4];
    const float* w2     = (const float*)d_in[5];
    const float* b2     = (const float*)d_in[6];
    const float* u2     = (const float*)d_in[7];
    const float* w3     = (const float*)d_in[8];
    const float* b3     = (const float*)d_in[9];
    const float* u3     = (const float*)d_in[10];
    const float* emb    = (const float*)d_in[11];
    const float* cond_w = (const float*)d_in[12];
    const float* cond_b = (const float*)d_in[13];
    const float* mb_w   = (const float*)d_in[14];
    const float* fc_w   = (const float*)d_in[15];
    const float* fc_b   = (const float*)d_in[16];
    float* out = (float*)d_out;

    char* ws = (char*)d_ws;
    size_t off = 0;
    auto carve = [&](size_t bytes) -> char* {
        char* p = ws + off;
        off += (bytes + 255) & ~(size_t)255;
        return p;
    };
    float*          scales = (float*)         carve(3 * sizeof(float));
    float*          x1     = (float*)         carve((size_t)NB*C1*T1 * sizeof(float));
    unsigned short* im2    = (unsigned short*)carve((size_t)M2*K2   * sizeof(unsigned short));
    float*          x2     = (float*)         carve((size_t)NB*C2*T2 * sizeof(float));
    unsigned short* im3    = (unsigned short*)carve((size_t)M3*K3   * sizeof(unsigned short));
    float*          flat   = (float*)         carve((size_t)NB*FLATD * sizeof(float));
    unsigned short* flatb  = (unsigned short*)carve((size_t)NB*FLATD * sizeof(unsigned short));
    unsigned short* w2b    = (unsigned short*)carve((size_t)C2*K2   * sizeof(unsigned short));
    unsigned short* w3b    = (unsigned short*)carve((size_t)C3*K3   * sizeof(unsigned short));
    unsigned short* mbwb   = (unsigned short*)carve((size_t)NMB_P*FLATD * sizeof(unsigned short));
    float*          act    = (float*)         carve((size_t)NB*NMB_R * sizeof(float));
    float*          feats  = (float*)         carve((size_t)NB*100  * sizeof(float));
    float*          condf  = (float*)         carve((size_t)NB*50   * sizeof(float));

    // 1) spectral-norm scales
    sn_sigma_kernel<<<1, 256, 0, stream>>>(w1, u1, 64,  15,  scales + 0);
    sn_sigma_kernel<<<1, 256, 0, stream>>>(w2, u2, 128, K2,  scales + 1);
    sn_sigma_kernel<<<1, 256, 0, stream>>>(w3, u3, 256, K3,  scales + 2);

    // 2) conv1 (scalar fp32)
    {
        long long total = (long long)NB * C1 * T1;
        conv1_kernel<<<(unsigned)((total + 255) / 256), 256, 0, stream>>>(ecg, w1, b1, scales + 0, x1);
    }

    // 3) bf16 weights (sigma-scaled); already N x K layout
    wcvt_kernel<<<(C2*K2 + 255) / 256, 256, 0, stream>>>(w2, w2b, scales + 1, C2*K2);
    wcvt_kernel<<<(C3*K3 + 255) / 256, 256, 0, stream>>>(w3, w3b, scales + 2, C3*K3);

    // 4) conv2 as WMMA GEMM  (M=82432, K=448, N=128)
    {
        long long total = (long long)M2 * K2;
        im2col_kernel<<<(unsigned)((total + 255) / 256), 256, 0, stream>>>(x1, im2, C1, T1, T2, 7, 3);
        dim3 grid(C2 / 64, M2 / 128);
        gemm_bf16_wmma_kernel<<<grid, 128, 0, stream>>>(im2, w2b, b2, x2,
                                                        M2, K2, C2, 1, T2, 0, C2*T2);
    }

    // 5) conv3 as WMMA GEMM (M=40448, K=1152, N=256); epilogue writes straight into flat
    {
        long long total = (long long)M3 * K3;
        im2col_kernel<<<(unsigned)((total + 255) / 256), 256, 0, stream>>>(x2, im3, C2, T2, T3, 9, 2);
        dim3 grid(C3 / 64, M3 / 128);
        gemm_bf16_wmma_kernel<<<grid, 128, 0, stream>>>(im3, w3b, b3, flat,
                                                        M3, K3, C3, 1, T3, 0, FLATD);
    }

    // 6) minibatch GEMM: act = flat @ mb_w   (M=512, K=20224, N padded 500->512)
    {
        long long total = (long long)NB * FLATD;
        cvt_bf16_kernel<<<(unsigned)((total + 255) / 256), 256, 0, stream>>>(flat, flatb, total);
        long long totalB = (long long)NMB_P * FLATD;
        mbwt_kernel<<<(unsigned)((totalB + 255) / 256), 256, 0, stream>>>(mb_w, mbwb);
        dim3 grid(NMB_P / 64, NB / 128);
        gemm_bf16_wmma_kernel<<<grid, 128, 0, stream>>>(flatb, mbwb, nullptr, act,
                                                        NB, FLATD, NMB_R, 0, 0, NMB_R, 0);
    }

    // 7) minibatch features, condition branch, final fc
    feats_kernel<<<(NB*100 + 255) / 256, 256, 0, stream>>>(act, feats);
    cond_kernel<<<(NB*50 + 255) / 256, 256, 0, stream>>>(condin, emb, cond_w, cond_b, condf);
    fc_kernel<<<NB, 256, 0, stream>>>(flat, feats, condf, fc_w, fc_b, out);
}
